// svsLanguage_classic_deterministic_46978352283808
// MI455X (gfx1250) — matrix operand order repaired and checked by reading, not compiled, not run on hardware
//
#include <hip/hip_runtime.h>
#include <hip/hip_bf16.h>

// ---------------------------------------------------------------------------
// LSTM autoencoder for MI455X (gfx1250), wave32 + WMMA bf16 (fp32 accumulate).
// gates(t) = x_t @ Wih^T + h @ Whh^T.  Each wave computes a 32(M) x 64(N)
// output strip: 2 A-fragments x 4 B-fragments -> 8 independent
// v_wmma_f32_16x16x32_bf16 per k-block (~512 FLOP/byte of L2 traffic).
// ---------------------------------------------------------------------------

typedef __attribute__((ext_vector_type(16))) __bf16 v16bf;
typedef __attribute__((ext_vector_type(8)))  float  v8f;

#define BATCH   1024
#define MTILES  (BATCH / 16)

// A-fragment (16x32 bf16): lane 0-15 row M=lane, K={0..7}+{16..23};
// lane 16-31 row M=lane-16, K={8..15}+{24..31}.  (ISA 7.12.2, 16-bit A 16x32)
// B-fragment (32x16 bf16): lanes 0-15 hold K={0..7}+{16..23}, lanes 16-31 hold K={8..15}+{24..31},
// column N = lane%16.  Weights stored row-major W[n][k] -> contiguous loads.
__device__ __forceinline__ void accum_strip(v8f acc[8],
                                            const float* __restrict__ A,
                                            const __bf16* __restrict__ B,
                                            int K, int row0, int col0,
                                            int half_, int m) {
    const float* arow0 = A + (size_t)(row0 + m) * K;
    const float* arow1 = arow0 + (size_t)16 * K;
    for (int k0 = 0; k0 < K; k0 += 32) {
        v16bf a0, a1;
#pragma unroll
        for (int j = 0; j < 8; ++j) {
            a0[j]     = (__bf16)arow0[k0 + half_ * 8 + j];
            a0[j + 8] = (__bf16)arow0[k0 + 16 + half_ * 8 + j];
            a1[j]     = (__bf16)arow1[k0 + half_ * 8 + j];
            a1[j + 8] = (__bf16)arow1[k0 + 16 + half_ * 8 + j];
        }
#pragma unroll
        for (int nt = 0; nt < 4; ++nt) {
            const __bf16* brow = B + (size_t)(col0 + nt * 16 + m) * K + half_ * 8 + k0;
            v16bf b;
#pragma unroll
            for (int j = 0; j < 16; ++j) b[j] = brow[(j & 7) + ((j >> 3) << 4)];
            acc[nt] = __builtin_amdgcn_wmma_f32_16x16x32_bf16(
                false, a0, false, b, (short)0, acc[nt], false, false);
            acc[4 + nt] = __builtin_amdgcn_wmma_f32_16x16x32_bf16(
                false, a1, false, b, (short)0, acc[4 + nt], false, false);
        }
    }
}

// C[b][n] = sum_k A1[b][k] W1[n][k]  (+ sum_k A2[b][k] W2[n][k]) , M = 1024.
// One wave computes one 32x64 strip. Npad is a multiple of 64; K1/K2 are
// padded to multiples of 32 and equal the row strides of A/B (pads hold 0).
__global__ void gates_gemm(const float* __restrict__ A1, const __bf16* __restrict__ B1, int K1,
                           const float* __restrict__ A2, const __bf16* __restrict__ B2, int K2,
                           float* __restrict__ C, int Npad) {
    const int wave  = threadIdx.x >> 5;
    const int lane  = threadIdx.x & 31;
    const int nsn   = Npad >> 6;                   // strips of 64 along N
    const int strip = blockIdx.x * (blockDim.x >> 5) + wave;
    if (strip >= nsn * (MTILES / 2)) return;       // whole-wave guard: EXEC all-ones
    const int sn = strip % nsn, sm = strip / nsn;
    const int row0 = sm << 5, col0 = sn << 6;
    const int half_ = lane >> 4, m = lane & 15;

    v8f acc[8];
#pragma unroll
    for (int i = 0; i < 8; ++i) acc[i] = {};

    accum_strip(acc, A1, B1, K1, row0, col0, half_, m);
    if (A2) accum_strip(acc, A2, B2, K2, row0, col0, half_, m);

    // D layout: VGPR r -> row +r + 8*(lane/16), col = lane%16
#pragma unroll
    for (int mt = 0; mt < 2; ++mt) {
#pragma unroll
        for (int nt = 0; nt < 4; ++nt) {
            float* crow = C + (size_t)(row0 + mt * 16 + half_ * 8) * Npad + col0 + nt * 16 + m;
#pragma unroll
            for (int r = 0; r < 8; ++r) crow[(size_t)r * Npad] = acc[mt * 4 + nt][r];
        }
    }
}

// ---------------------------------------------------------------------------
__device__ __forceinline__ float sigf(float x) { return 1.0f / (1.0f + __expf(-x)); }

// LSTM cell pointwise: gate order i,f,g,o (PyTorch). mode 0: none,
// mode 1: extra = relu(h), mode 2: extra = h (per-timestep output copy).
__global__ void lstm_pointwise(const float* __restrict__ G, int ldG,
                               const float* __restrict__ bih, const float* __restrict__ bhh,
                               float* __restrict__ h, float* __restrict__ c, int ldS, int H,
                               float* __restrict__ extra, int ldE, int mode) {
    int idx = blockIdx.x * blockDim.x + threadIdx.x;
    if (idx >= BATCH * H) return;
    int b = idx / H, j = idx - b * H;
    const float* g = G + (size_t)b * ldG;
    float xi = g[j]         + bih[j]         + bhh[j];
    float xf = g[H + j]     + bih[H + j]     + bhh[H + j];
    float xg = g[2 * H + j] + bih[2 * H + j] + bhh[2 * H + j];
    float xo = g[3 * H + j] + bih[3 * H + j] + bhh[3 * H + j];
    float cn = sigf(xf) * c[(size_t)b * ldS + j] + sigf(xi) * tanhf(xg);
    float hn = sigf(xo) * tanhf(cn);
    c[(size_t)b * ldS + j] = cn;
    h[(size_t)b * ldS + j] = hn;
    if (mode == 1)      extra[(size_t)b * ldE + j] = fmaxf(hn, 0.0f);
    else if (mode == 2) extra[(size_t)b * ldE + j] = hn;
}

__global__ void linear_bias(const float* __restrict__ G, int ldG,
                            const float* __restrict__ bias,
                            float* __restrict__ out, int ldO, int H) {
    int idx = blockIdx.x * blockDim.x + threadIdx.x;
    if (idx >= BATCH * H) return;
    int b = idx / H, j = idx - b * H;
    out[(size_t)b * ldO + j] = G[(size_t)b * ldG + j] + bias[j];
}

// fp32 weight [rows][cols] -> zero-padded bf16 [rpad][cpad]
__global__ void cvt_pad(const float* __restrict__ W, int rows, int cols,
                        __bf16* __restrict__ O, int rpad, int cpad) {
    int idx = blockIdx.x * blockDim.x + threadIdx.x;
    if (idx >= rpad * cpad) return;
    int r = idx / cpad, cc = idx - r * cpad;
    float v = (r < rows && cc < cols) ? W[(size_t)r * cols + cc] : 0.0f;
    O[idx] = (__bf16)v;
}

__global__ void copy_rows(const float* __restrict__ S, int ldS,
                          float* __restrict__ O, int H) {
    int idx = blockIdx.x * blockDim.x + threadIdx.x;
    if (idx >= BATCH * H) return;
    int b = idx / H, j = idx - b * H;
    O[(size_t)b * H + j] = S[(size_t)b * ldS + j];
}

// ---------------------------------------------------------------------------
extern "C" void kernel_launch(void* const* d_in, const int* in_sizes, int n_in,
                              void* d_out, int out_size, void* d_ws, size_t ws_size,
                              hipStream_t stream) {
    const float* x        = (const float*)d_in[0];
    const float* fc1_Wih  = (const float*)d_in[1];
    const float* fc1_Whh  = (const float*)d_in[2];
    const float* fc1_bih  = (const float*)d_in[3];
    const float* fc1_bhh  = (const float*)d_in[4];
    const float* fc21_Wih = (const float*)d_in[5];
    const float* fc21_Whh = (const float*)d_in[6];
    const float* fc21_bih = (const float*)d_in[7];
    const float* fc21_bhh = (const float*)d_in[8];
    const float* l3_Wih   = (const float*)d_in[9];
    const float* l3_Whh   = (const float*)d_in[10];
    const float* l3_bih   = (const float*)d_in[11];
    const float* l3_bhh   = (const float*)d_in[12];
    const float* l4_Wih   = (const float*)d_in[13];
    const float* l4_Whh   = (const float*)d_in[14];
    const float* l4_bih   = (const float*)d_in[15];
    const float* l4_bhh   = (const float*)d_in[16];
    const float* l5_Wih   = (const float*)d_in[17];
    const float* l5_Whh   = (const float*)d_in[18];
    const float* l5_bih   = (const float*)d_in[19];
    const float* l5_bhh   = (const float*)d_in[20];
    const float* id_W     = (const float*)d_in[21];
    const float* id_b     = (const float*)d_in[22];
    const float* down_W   = (const float*)d_in[23];
    const float* down_b   = (const float*)d_in[24];
    float* out = (float*)d_out;

    // ---- workspace carve-up (256B aligned) ----
    char* ws = (char*)d_ws;
    size_t off = 0;
    auto take = [&](size_t bytes) { size_t o = off; off += (bytes + 255) & ~(size_t)255; return o; };

    // N padded to multiples of 64 (zero rows), K padded to multiples of 32.
    __bf16* w_fc1_ih  = (__bf16*)(ws + take(256  * 128 * 2));   // N 240->256
    __bf16* w_fc1_hh  = (__bf16*)(ws + take(256  * 64  * 2));   // K 60->64
    __bf16* w_fc21_ih = (__bf16*)(ws + take(128  * 64  * 2));   // N 80->128
    __bf16* w_fc21_hh = (__bf16*)(ws + take(128  * 32  * 2));   // K 20->32
    __bf16* w_l3_ih   = (__bf16*)(ws + take(128  * 32  * 2));
    __bf16* w_l3_hh   = (__bf16*)(ws + take(128  * 32  * 2));
    __bf16* w_l4_ih   = (__bf16*)(ws + take(3200 * 32  * 2));
    __bf16* w_l4_hh   = (__bf16*)(ws + take((size_t)3200 * 800 * 2));
    __bf16* w_l5_ih   = (__bf16*)(ws + take((size_t)512 * 800 * 2));
    __bf16* w_l5_hh   = (__bf16*)(ws + take(512 * 128 * 2));
    __bf16* w_id      = (__bf16*)(ws + take(64 * 32 * 2));      // N 20->64, K 20->32
    __bf16* w_down    = (__bf16*)(ws + take(64 * 128 * 2));     // N 20->64

    size_t zbase = off;                                         // zeroed fp32 region
    float* h1    = (float*)(ws + take((size_t)BATCH * 64  * 4));
    float* c1    = (float*)(ws + take((size_t)BATCH * 64  * 4));
    float* relu1 = (float*)(ws + take((size_t)BATCH * 64  * 4));
    float* h21   = (float*)(ws + take((size_t)BATCH * 32  * 4));
    float* c21   = (float*)(ws + take((size_t)BATCH * 32  * 4));
    float* h3s   = (float*)(ws + take((size_t)BATCH * 32  * 4));
    float* c3s   = (float*)(ws + take((size_t)BATCH * 32  * 4));
    float* h4    = (float*)(ws + take((size_t)BATCH * 800 * 4));
    float* c4    = (float*)(ws + take((size_t)BATCH * 800 * 4));
    float* h5    = (float*)(ws + take((size_t)BATCH * 128 * 4));
    float* c5    = (float*)(ws + take((size_t)BATCH * 128 * 4));
    float* outb  = (float*)(ws + take((size_t)BATCH * 32  * 4));
    float* gates = (float*)(ws + take((size_t)BATCH * 3200 * 4));
    size_t zsize = off - zbase;
    (void)ws_size; (void)in_sizes; (void)n_in; (void)out_size;

    // zero all states + padding lanes every call (h0=c0=0, pads stay 0)
    hipMemsetAsync(ws + zbase, 0, zsize, stream);

    auto cvt = [&](const float* W, int r, int c, __bf16* O, int rp, int cp) {
        int n = rp * cp;
        cvt_pad<<<(n + 255) / 256, 256, 0, stream>>>(W, r, c, O, rp, cp);
    };
    cvt(fc1_Wih, 240, 128, w_fc1_ih, 256, 128);
    cvt(fc1_Whh, 240, 60,  w_fc1_hh, 256, 64);
    cvt(fc21_Wih, 80, 60,  w_fc21_ih, 128, 64);
    cvt(fc21_Whh, 80, 20,  w_fc21_hh, 128, 32);
    cvt(l3_Wih,   80, 20,  w_l3_ih,   128, 32);
    cvt(l3_Whh,   80, 20,  w_l3_hh,   128, 32);
    cvt(l4_Wih, 3200, 20,  w_l4_ih, 3200, 32);
    cvt(l4_Whh, 3200, 800, w_l4_hh, 3200, 800);
    cvt(l5_Wih,  512, 800, w_l5_ih,  512, 800);
    cvt(l5_Whh,  512, 128, w_l5_hh,  512, 128);
    cvt(id_W,     20, 20,  w_id,      64, 32);
    cvt(down_W,   20, 128, w_down,    64, 128);

    auto gemm = [&](const float* A1, const __bf16* B1, int K1,
                    const float* A2, const __bf16* B2, int K2,
                    float* C, int Npad) {
        int strips = (Npad >> 6) * (MTILES / 2);   // 32x64 strips
        gates_gemm<<<(strips + 7) / 8, 256, 0, stream>>>(A1, B1, K1, A2, B2, K2, C, Npad);
    };
    auto pw = [&](const float* G, int ldG, const float* bi, const float* bh,
                  float* h, float* c, int ldS, int H, float* ex, int ldE, int mode) {
        int n = BATCH * H;
        lstm_pointwise<<<(n + 255) / 256, 256, 0, stream>>>(G, ldG, bi, bh, h, c, ldS, H, ex, ldE, mode);
    };

    // ---------------- encoder: 256 steps of fc1 (128->60) then fc21 (60->20) ----
    for (int t = 0; t < 256; ++t) {
        const float* xt = x + (size_t)t * BATCH * 128;
        gemm(xt, w_fc1_ih, 128, h1, w_fc1_hh, 64, gates, 256);
        pw(gates, 256, fc1_bih, fc1_bhh, h1, c1, 64, 60, relu1, 64, /*relu out*/1);
        gemm(relu1, w_fc21_ih, 64, h21, w_fc21_hh, 32, gates, 128);
        pw(gates, 128, fc21_bih, fc21_bhh, h21, c21, 32, 20, nullptr, 0, 0);
    }

    // decoder initial state: h=mu (=h21 final), c=c_mean (=c21 final)
    hipMemcpyAsync(h3s, h21, (size_t)BATCH * 32 * 4, hipMemcpyDeviceToDevice, stream);
    hipMemcpyAsync(c3s, c21, (size_t)BATCH * 32 * 4, hipMemcpyDeviceToDevice, stream);

    // output0 = mu @ id_W^T + id_b
    gemm(h21, w_id, 32, nullptr, nullptr, 0, gates, 64);
    linear_bias<<<(BATCH * 20 + 255) / 256, 256, 0, stream>>>(gates, 64, id_b, outb, 32, 20);

    // ---------------- decoder: 256 steps of l3 (20->20), l4 (20->800), l5 (800->128)
    for (int t = 0; t < 256; ++t) {
        gemm(outb, w_l3_ih, 32, h3s, w_l3_hh, 32, gates, 128);
        pw(gates, 128, l3_bih, l3_bhh, h3s, c3s, 32, 20, nullptr, 0, 0);

        gemm(h3s, w_l4_ih, 32, h4, w_l4_hh, 800, gates, 3200);
        pw(gates, 3200, l4_bih, l4_bhh, h4, c4, 800, 800, nullptr, 0, 0);

        float* out_t = out + (size_t)t * BATCH * 128;
        gemm(h4, w_l5_ih, 800, h5, w_l5_hh, 128, gates, 512);
        pw(gates, 512, l5_bih, l5_bhh, h5, c5, 128, 128, out_t, 128, /*copy h*/2);

        gemm(h5, w_down, 128, nullptr, nullptr, 0, gates, 64);
        linear_bias<<<(BATCH * 20 + 255) / 256, 256, 0, stream>>>(gates, 64, down_b, outb, 32, 20);
    }

    // append mu (1024 x 20) after outMean (256 x 1024 x 128)
    copy_rows<<<(BATCH * 20 + 255) / 256, 256, 0, stream>>>(h21, 32, out + (size_t)256 * BATCH * 128, 20);
}
